// SimpleTreeLSTMCell_66571993088453
// MI455X (gfx1250) — compile-verified, matrix operand order checked by reading, not run on hardware
//
#include <hip/hip_runtime.h>
#include <hip/hip_bf16.h>
#include <math.h>

// Problem constants (from reference): T=B=I=H=256
#define T_STEPS 256
#define BB      256
#define II      256
#define HH      256
#define NTOT    1024   // 4*H
#define KTOT    768    // I + H (parent) + H (y_{t-1})
#define KT_TILES 24    // KTOT / 32
#define NT_TILES 64    // NTOT / 16
#define ROWS    16     // batch rows per workgroup
#define NBLK    (BB / ROWS)
#define SLAB_KT 2      // K-tiles per TDM slab
#define SLABS   (KT_TILES / SLAB_KT)
#define SLAB_BYTES (SLAB_KT * NT_TILES * 1024)   // 128 KB

typedef __attribute__((ext_vector_type(16))) __bf16 v16bf;
typedef __attribute__((ext_vector_type(8)))  float  v8f;
typedef __attribute__((ext_vector_type(4)))  unsigned int v4u;
typedef __attribute__((ext_vector_type(8)))  int  v8i;
typedef __attribute__((ext_vector_type(4)))  int  v4i;

struct __align__(16) U4 { unsigned int x, y, z, w; };
union Frag16 { v16bf v; U4 q[2]; };

// ---- CDNA5 TDM availability (device pass only; host pass takes fallback) ----
#if defined(__HIP_DEVICE_COMPILE__) && __has_builtin(__builtin_amdgcn_tensor_load_to_lds)
#define USE_TDM 1
#else
#define USE_TDM 0
#endif

#if defined(__HIP_DEVICE_COMPILE__) && __has_builtin(__builtin_amdgcn_s_wait_tensorcnt)
#define WAIT_TENSOR0() __builtin_amdgcn_s_wait_tensorcnt(0)
#else
#define WAIT_TENSOR0() asm volatile("s_wait_tensorcnt 0x0" ::: "memory")
#endif

#if USE_TDM
// One TDM transfer: contiguous n8 x 8-byte elements, global -> LDS.
// D# bitfields per CDNA5 ISA 8.3-8.6:
//  group0: [1:0] count=1, [63:32] lds_addr, [120:64] global_addr, [127:126] type=2
//  group1: [17:16] data_size=3 (8B), [79:48] tensor_dim0, [111:80] tensor_dim1,
//          [127:112] tile_dim0, [143:128] tile_dim1=0, [207:160] tensor_dim0_stride
__device__ __forceinline__ void tdm_load_slab(const void* gsrc, unsigned lds_addr,
                                              unsigned n8) {
  unsigned long long ga = (unsigned long long)(uintptr_t)gsrc;
  v4u g0 = { 1u,                                   // count=1 (valid user desc)
             lds_addr,                             // LDS byte address
             (unsigned)(ga & 0xFFFFFFFFu),         // global_addr[31:0]
             (unsigned)((ga >> 32) & 0x01FFFFFFu) | 0x80000000u };  // [56:32]|type=2
  union { v8i v; int i[8]; } g1 = {};
  g1.i[0] = (int)(3u << 16);                       // data_size = 8 bytes
  g1.i[1] = (int)((n8 & 0xFFFFu) << 16);           // tensor_dim0[15:0] -> [63:48]
  g1.i[2] = (int)(((n8 >> 16) & 0xFFFFu) | (1u << 16)); // dim0[31:16], tensor_dim1=1
  g1.i[3] = (int)((n8 & 0xFFFFu) << 16);           // tile_dim0 -> [127:112]
  g1.i[4] = 0;                                     // tile_dim1=0, tile_dim2=0
  g1.i[5] = (int)n8;                               // tensor_dim0_stride[31:0]
  g1.i[6] = 0;
  g1.i[7] = 0;
  v4i gz = { 0, 0, 0, 0 };
#if __clang_major__ >= 23
  v8i gz8 = {};
  __builtin_amdgcn_tensor_load_to_lds(g0, g1.v, gz, gz, gz8, 0);
#else
  __builtin_amdgcn_tensor_load_to_lds(g0, g1.v, gz, gz, 0);
#endif
}
#endif  // USE_TDM

// Gate-interleaved n-tile ownership: wave w, slot i (0..7) ->
// gate g = i>>1, pair p = i&1, n-tile = g*16 + w*2 + p.
// A wave's 8 accumulators hold matching (ig,fg,og,u) columns for
// h in [32w, 32w+32) -> the whole LSTM cell update stays in registers.
__device__ __forceinline__ constexpr int ntile_of(int w, int i) {
  return ((i >> 1) << 4) + w * 2 + (i & 1);
}

// ---------------------------------------------------------------------------
// Pre-pass: pack [W_ih | W_hh] (fp32) into bf16 WMMA-B blobs, kt-MAJOR so a
// K-slab (all 64 n-tiles for SLAB_KT kt's) is one contiguous TDM transfer.
// Blob (kt, nt) at (kt*64+nt)*1KB. Lane L: n = nt*16 + (L&15);
// first 8 bf16 = K kt*32+(L>=16?8:0)..+7, second 8 = same + 16.
// ---------------------------------------------------------------------------
__global__ __launch_bounds__(256) void pack_weights_kernel(
    const float* __restrict__ wih,   // (1024, 512)
    const float* __restrict__ whh,   // (1024, 256)
    __bf16* __restrict__ wpack)      // 1536 blobs * 512 bf16 = 1.5 MB
{
  int idx = blockIdx.x * 256 + threadIdx.x;       // 0 .. 786431
  if (idx >= NT_TILES * KT_TILES * 512) return;
  int blob = idx >> 9;                            // 512 elements per blob
  int e    = idx & 511;
  int lane = e >> 4;                              // 0..31
  int slot = e & 15;                              // 0..15
  int kt = blob >> 6;                             // kt-major
  int nt = blob & 63;
  int n  = nt * 16 + (lane & 15);
  int kb = kt * 32 + ((lane >> 4) * 8);
  int k  = kb + ((slot < 8) ? slot : (slot - 8 + 16));
  float v = (k < 512) ? wih[n * 512 + k] : whh[n * 256 + (k - 512)];
  wpack[(size_t)blob * 512 + e] = (__bf16)v;
}

// ---------------------------------------------------------------------------
// Persistent TreeLSTM: one workgroup = 16 batch rows, loops over all 256 steps.
// Weights TDM-staged into a double-buffered 2x128KB LDS slab (2 K-tiles each).
// Gates + cell state fully register-resident via gate-interleaved tiling.
// ---------------------------------------------------------------------------
__global__ __launch_bounds__(256) void tree_lstm_kernel(
    const float* __restrict__ x,        // (T, B, I)
    const int*   __restrict__ pushpop,  // (T, B)
    const __bf16* __restrict__ wpack,   // packed B blobs (kt-major)
    const float* __restrict__ bias,     // (4H)
    float*       __restrict__ ys)       // (T, B, H) output
{
  __shared__ __align__(16) __bf16 sA[ROWS][KTOT];     // 24 KB  A staging (bf16)
  __shared__ int   sStack[ROWS][T_STEPS + 4];         // ~16.6 KB pointer stacks
  __shared__ int   sDepth[ROWS];
  __shared__ int   sPtr[ROWS];
#if USE_TDM
  __shared__ __align__(128) char sB[2][SLAB_BYTES];   // 256 KB double-buffer
#endif

  const int tid  = threadIdx.x;
  const int lane = tid & 31;
  const int wave = tid >> 5;              // 0..7
  const int b0   = blockIdx.x * ROWS;

  const int laneRow = lane & 15;          // M (A frags) / N-col within tile
  const int laneHi  = lane >> 4;          // K-half select (A), M-half (D)

  const int erow  = tid >> 4;             // staging row 0..15
  const int ecol0 = (tid & 15) * 16;      // 16 columns per thread

  // ---- init persistent LDS + register state ----
  for (int i = tid; i < ROWS * (T_STEPS + 4); i += 256) ((int*)sStack)[i] = 0;
  for (int i = tid; i < ROWS * HH; i += 256) {
    int r = i >> 8, h = i & 255;
    sA[r][512 + h] = (__bf16)0.0f;        // y_{-1} = 0
  }
  if (tid < ROWS) sDepth[tid] = 1;

  // Per-lane bias (constant across t): gate g, pair p -> h = 32w + 16p + laneRow
  const int h0 = wave * 32 + laneRow;
  float bgate[4][2];
  #pragma unroll
  for (int g = 0; g < 4; ++g) {
    bgate[g][0] = bias[g * 256 + h0];
    bgate[g][1] = bias[g * 256 + h0 + 16];
  }
  float creg[2][8] = {};                  // cell state, D-matrix layout
  __syncthreads();

  const char* wbase = (const char*)wpack;
#if USE_TDM
  const unsigned sb_addr[2] = { (unsigned)(uintptr_t)(void*)&sB[0][0],
                                (unsigned)(uintptr_t)(void*)&sB[1][0] };
#endif

  for (int t = 0; t < T_STEPS; ++t) {
#if USE_TDM
    // Prefetch slab 0; overlaps with A staging. Safe: end-of-step barrier
    // guarantees all waves finished reading both buffers.
    if (wave == 0) tdm_load_slab(wbase, sb_addr[0], SLAB_BYTES / 8);
#endif
    // ---- top-of-stack pointer per row ----
    if (tid < ROWS) sPtr[tid] = sStack[tid][sDepth[tid] - 1];
    __syncthreads();

    // ---- stage A: x_t (cols 0..255) and parent_y (cols 256..511) as bf16 ----
    {
      const size_t xbase = ((size_t)t * BB + b0 + erow) * II + ecol0;
      #pragma unroll
      for (int j = 0; j < 16; ++j)
        sA[erow][ecol0 + j] = (__bf16)x[xbase + j];

      const int ptr = sPtr[erow];
      if (ptr >= 2) {                    // prev_ys[b][k] == ys[k-2][b]
        const size_t pbase = ((size_t)(ptr - 2) * BB + b0 + erow) * HH + ecol0;
        #pragma unroll
        for (int j = 0; j < 16; ++j)
          sA[erow][256 + ecol0 + j] = (__bf16)ys[pbase + j];
      } else {
        #pragma unroll
        for (int j = 0; j < 16; ++j)
          sA[erow][256 + ecol0 + j] = (__bf16)0.0f;
      }
    }
    __syncthreads();

    // ---- GEMM: acts(16x1024) = A(16x768) x B(768x1024), bf16 WMMA ----
    v8f acc[8] = {};
#if USE_TDM
    #pragma unroll 1
    for (int s = 0; s < SLABS; ++s) {
      const int cur = s & 1;
      if (wave == 0) WAIT_TENSOR0();                // slab s landed in LDS
      __syncthreads();                              // publish slab / retire reads
      if ((wave == 0) && (s + 1 < SLABS))           // async-pull next slab
        tdm_load_slab(wbase + ((size_t)(s + 1)) * SLAB_BYTES,
                      sb_addr[1 - cur], SLAB_BYTES / 8);
      // Per-K-tile base pointers: all blob offsets stay < 64 KB so every
      // ds_load_b128 gets an immediate offset (no per-load address VALU).
      const char* bp0 = &sB[cur][0] + lane * 32;    // kk = 0
      const char* bp1 = bp0 + NT_TILES * 1024;      // kk = 1
      // A fragments for the slab's two K-tiles
      Frag16 aF[SLAB_KT];
      #pragma unroll
      for (int kk = 0; kk < SLAB_KT; ++kk) {
        const int kb = (s * SLAB_KT + kk) * 32 + laneHi * 8;
        aF[kk].q[0] = *(const U4*)&sA[laneRow][kb];
        aF[kk].q[1] = *(const U4*)&sA[laneRow][kb + 16];
      }
      // Explicit register ping-pong (no rotating copy -> no coalescing):
      // even slot consumes bA while bB loads, odd slot consumes bB while
      // bA reloads. DScnt retires in order => s_wait_dscnt 0x2 per WMMA.
      Frag16 bA, bB;
      {
        const U4* p = (const U4*)(bp0 + (ntile_of(0, 0) << 10) + (wave << 11));
        // ntile_of(wave,0) = wave*2 -> offset wave*2048
        bA.q[0] = p[0]; bA.q[1] = p[1];
      }
      #pragma unroll
      for (int d2 = 0; d2 < SLAB_KT * 8; d2 += 2) {
        const int i0 = d2 & 7,        k0 = d2 >> 3;
        const int i1 = (d2 + 1) & 7,  k1 = (d2 + 1) >> 3;
        {  // load partner for the odd slot
          const int nn = ntile_of(wave, i1) << 10;
          const U4* p = (const U4*)((k1 ? bp1 : bp0) + nn);
          bB.q[0] = p[0]; bB.q[1] = p[1];
        }
        acc[i0] = __builtin_amdgcn_wmma_f32_16x16x32_bf16(
            false, aF[k0].v, false, bA.v, (short)0, acc[i0], false, false);
        if (d2 + 2 < SLAB_KT * 8) {   // reload bA for the next even slot
          const int i2 = (d2 + 2) & 7, k2 = (d2 + 2) >> 3;
          const int nn = ntile_of(wave, i2) << 10;
          const U4* p = (const U4*)((k2 ? bp1 : bp0) + nn);
          bA.q[0] = p[0]; bA.q[1] = p[1];
        }
        acc[i1] = __builtin_amdgcn_wmma_f32_16x16x32_bf16(
            false, aF[k1].v, false, bB.v, (short)0, acc[i1], false, false);
      }
    }
#else
    #pragma unroll 1
    for (int kt = 0; kt < KT_TILES; ++kt) {
      Frag16 a;
      const int kb = kt * 32 + laneHi * 8;
      a.q[0] = *(const U4*)&sA[laneRow][kb];
      a.q[1] = *(const U4*)&sA[laneRow][kb + 16];
      #pragma unroll
      for (int i = 0; i < 8; ++i) {
        const U4* bp = (const U4*)(wbase +
            (((size_t)(kt * NT_TILES + ntile_of(wave, i))) << 10)
            + (size_t)lane * 32);
        Frag16 b;
        b.q[0] = bp[0]; b.q[1] = bp[1];
        acc[i] = __builtin_amdgcn_wmma_f32_16x16x32_bf16(
            false, a.v, false, b.v, (short)0, acc[i], false, false);
      }
    }
#endif

    // ---- gates + cell update: fully register-resident ----
    // acc[g*2+p][j] = acts[row = j + 8*laneHi][h = 32w + 16p + laneRow] + gate g
    {
      #pragma unroll
      for (int p = 0; p < 2; ++p) {
        const int h = h0 + p * 16;
        #pragma unroll
        for (int j = 0; j < 8; ++j) {
          const int row = j + laneHi * 8;
          float ig = acc[0 + p][j] + bgate[0][p];
          float fg = acc[2 + p][j] + bgate[1][p];
          float og = acc[4 + p][j] + bgate[2][p];
          float u  = acc[6 + p][j] + bgate[3][p];
          ig = 1.0f / (1.0f + __expf(-ig));
          fg = 1.0f / (1.0f + __expf(-fg));
          og = 1.0f / (1.0f + __expf(-og));
          float ct = (1.0f - fg) * creg[p][j] + ig * tanhf(u);
          float yt = og * tanhf(ct);
          creg[p][j] = ct;
          ys[((size_t)t * BB + b0 + row) * HH + h] = yt;  // output / prev_ys
          sA[row][512 + h] = (__bf16)yt;                  // y_t -> next step A
        }
      }
    }

    // ---- pointer-stack bookkeeping (one lane per row) ----
    if (tid < ROWS) {
      const int r  = tid;
      const int pp = pushpop[t * BB + b0 + r];
      const int d  = sDepth[r];
      const int k  = t + 1;
      const int cur = sStack[r][d];
      sStack[r][d] = (pp > 0) ? k : cur;
      int dm1 = d - 1; if (dm1 < 1) dm1 = 1;
      sDepth[r] = (pp > 0) ? (d + 1) : ((pp < 0) ? dm1 : d);
    }
    __syncthreads();
  }
}

extern "C" void kernel_launch(void* const* d_in, const int* in_sizes, int n_in,
                              void* d_out, int out_size, void* d_ws, size_t ws_size,
                              hipStream_t stream) {
  const float* x      = (const float*)d_in[0];   // (T,B,I) fp32
  const int*   pp     = (const int*)  d_in[1];   // (T,B)   int32
  const float* wih    = (const float*)d_in[2];   // (1024,512) fp32
  const float* whh    = (const float*)d_in[3];   // (1024,256) fp32
  const float* bias   = (const float*)d_in[4];   // (1024) fp32
  float*       ys     = (float*)d_out;           // (T,B,H) fp32
  __bf16*      wpack  = (__bf16*)d_ws;           // 1.5 MB packed bf16 weights

  // 786432 elements -> 3072 blocks of 256
  pack_weights_kernel<<<3072, 256, 0, stream>>>(wih, whh, wpack);
  tree_lstm_kernel<<<NBLK, 256, 0, stream>>>(x, pp, wpack, bias, ys);
}
